// GNN_Decoder_52390011076802
// MI455X (gfx1250) — compile-verified, matrix operand order checked by reading, not run on hardware
//
#include <hip/hip_runtime.h>
#include <hip/hip_bf16.h>
#include <stdint.h>

#define NEG_SLOPE 0.2f

typedef __attribute__((ext_vector_type(16))) _Float16 v16h;
typedef __attribute__((ext_vector_type(8)))  float    v8f;

__device__ __forceinline__ v8f wmma_f16(v16h a, v16h b, v8f c) {
  // 8 args: (neg_a, A, neg_b, B, c_mod, C, reuse_a, reuse_b)
  return __builtin_amdgcn_wmma_f32_16x16x32_f16(false, a, false, b, (short)0, c, false, false);
}

// ---- ordered-uint encoding for float atomic max --------------------------
__device__ __forceinline__ unsigned f2ord(float f) {
  unsigned u = __float_as_uint(f);
  return (u & 0x80000000u) ? ~u : (u | 0x80000000u);
}
__device__ __forceinline__ float ord2f(unsigned u) {
  return (u & 0x80000000u) ? __uint_as_float(u & 0x7FFFFFFFu) : __uint_as_float(~u);
}
#define ORD_NEG_INF 0x007FFFFFu  // f2ord(-inf)

// Edge (src,dst) for both layers: e < E -> original edge; otherwise self-loop
// (layer1 appends N loops; layer2 appends N more on top of layer1's list).
__device__ __forceinline__ void edge_sd(const int* __restrict__ ei, int E, int N,
                                        int e, int& s, int& d) {
  if (e < E) { s = ei[e]; d = ei[(size_t)E + e]; }
  else { int t = e - E; if (t >= N) t -= N; s = t; d = t; }
}

// ---- init kernels --------------------------------------------------------
__global__ void init1_k(unsigned* __restrict__ maxd, float* __restrict__ sumd,
                        float* __restrict__ out1, const float* __restrict__ b1,
                        float* __restrict__ scal, int N) {
  int i = blockIdx.x * blockDim.x + threadIdx.x;
  if (i >= N) return;
  maxd[i] = ORD_NEG_INF;
  sumd[i] = 0.f;
  #pragma unroll
  for (int c = 0; c < 16; ++c) out1[(size_t)i * 16 + c] = b1[c];
  if (i == 0) scal[0] = 0.f;
}

// relu layer-1 output in place, re-init segment arrays, init layer-2 output
__global__ void layer1_finish_k(float* __restrict__ h, unsigned* __restrict__ maxd,
                                float* __restrict__ sumd, float* __restrict__ out2,
                                const float* __restrict__ b2, float* __restrict__ scal,
                                int N) {
  int i = blockIdx.x * blockDim.x + threadIdx.x;
  if (i >= N) return;
  #pragma unroll
  for (int c = 0; c < 16; ++c) {
    float v = h[(size_t)i * 16 + c];
    h[(size_t)i * 16 + c] = v > 0.f ? v : 0.f;
  }
  maxd[i] = ORD_NEG_INF;
  sumd[i] = 0.f;
  #pragma unroll
  for (int c = 0; c < 32; ++c) out2[(size_t)i * 32 + c] = b2[c];
  if (i == 0) scal[1] = 0.f;
}

// ---- grid-stride sum reduction ------------------------------------------
__global__ void reduce_sum_k(const float* __restrict__ v, long long n,
                             float* __restrict__ out) {
  __shared__ float sm[256];
  float acc = 0.f;
  for (long long i = (long long)blockIdx.x * 256 + threadIdx.x; i < n;
       i += (long long)gridDim.x * 256)
    acc += v[i];
  sm[threadIdx.x] = acc;
  __syncthreads();
  for (int s = 128; s > 0; s >>= 1) {
    if (threadIdx.x < s) sm[threadIdx.x] += sm[threadIdx.x + s];
    __syncthreads();
  }
  if (threadIdx.x == 0) atomicAdd(out, sm[0]);
}

// ---- GEMM 1: xl1 = x@W1l, xr1 = x@W1r  ([N,128]x[128,16]) via WMMA -------
__global__ void gemm1_k(const float* __restrict__ x, const float* __restrict__ Wl,
                        const float* __restrict__ Wr, float* __restrict__ xl,
                        float* __restrict__ xr, int Ntiles, int N) {
  int wave = (int)((blockIdx.x * blockDim.x + threadIdx.x) >> 5);
  int lane = threadIdx.x & 31;
  if (wave >= Ntiles) return;
  int half = lane >> 4;          // 0: k 0-7 & 16-23, 1: k 8-15 & 24-31
  int kb   = half * 8;
  int ncol = lane & 15;
  int mrow = wave * 16 + (lane & 15);
  int mr   = mrow < N ? mrow : N - 1;
  const float* xrow = x + (size_t)mr * 128;

  v8f cl = {}, cr = {};
  #pragma unroll
  for (int ks = 0; ks < 4; ++ks) {
    int k0 = ks * 32 + kb;
    v16h a, bl, br;
    #pragma unroll
    for (int i = 0; i < 8; ++i) {
      a[i]     = (_Float16)xrow[k0 + i];
      a[8 + i] = (_Float16)xrow[k0 + 16 + i];
      bl[i]     = (_Float16)Wl[(k0 + i) * 16 + ncol];
      bl[8 + i] = (_Float16)Wl[(k0 + 16 + i) * 16 + ncol];
      br[i]     = (_Float16)Wr[(k0 + i) * 16 + ncol];
      br[8 + i] = (_Float16)Wr[(k0 + 16 + i) * 16 + ncol];
    }
    cl = wmma_f16(a, bl, cl);
    cr = wmma_f16(a, br, cr);
  }
  int rbase = wave * 16 + half * 8;
  #pragma unroll
  for (int j = 0; j < 8; ++j) {
    int r = rbase + j;
    if (r < N) {
      xl[(size_t)r * 16 + ncol] = cl[j];
      xr[(size_t)r * 16 + ncol] = cr[j];
    }
  }
}

// ---- GEMM 2: xl2 = h@W2l, xr2 = h@W2r  ([N,16]x[16,32], K zero-padded) ---
__global__ void gemm2_k(const float* __restrict__ h, const float* __restrict__ Wl,
                        const float* __restrict__ Wr, float* __restrict__ xl,
                        float* __restrict__ xr, int Ntiles, int N) {
  int wave = (int)((blockIdx.x * blockDim.x + threadIdx.x) >> 5);
  int lane = threadIdx.x & 31;
  if (wave >= Ntiles) return;
  int half = lane >> 4;
  int kb   = half * 8;
  int ncol = lane & 15;
  int mrow = wave * 16 + (lane & 15);
  int mr   = mrow < N ? mrow : N - 1;
  const float* hrow = h + (size_t)mr * 16;

  v16h a = {};   // K = 16..31 stays zero (K padded to 32)
  #pragma unroll
  for (int i = 0; i < 8; ++i) a[i] = (_Float16)hrow[kb + i];

  v8f accL[2] = {{}, {}};
  v8f accR[2] = {{}, {}};
  #pragma unroll
  for (int nt = 0; nt < 2; ++nt) {
    v16h bl = {}, br = {};
    #pragma unroll
    for (int i = 0; i < 8; ++i) {
      bl[i] = (_Float16)Wl[(kb + i) * 32 + nt * 16 + ncol];
      br[i] = (_Float16)Wr[(kb + i) * 32 + nt * 16 + ncol];
    }
    accL[nt] = wmma_f16(a, bl, accL[nt]);
    accR[nt] = wmma_f16(a, br, accR[nt]);
  }
  int rbase = wave * 16 + half * 8;
  #pragma unroll
  for (int nt = 0; nt < 2; ++nt)
    #pragma unroll
    for (int j = 0; j < 8; ++j) {
      int r = rbase + j;
      if (r < N) {
        xl[(size_t)r * 32 + nt * 16 + ncol] = accL[nt][j];
        xr[(size_t)r * 32 + nt * 16 + ncol] = accR[nt][j];
      }
    }
}

// ---- edge pass 1: logits + per-dst max -----------------------------------
template <int C, int LAYER>
__global__ void edge_logits_k(const int* __restrict__ ei, int E, int N, int Etot,
                              const float* __restrict__ eaSrc,  // ea (L1) / alpha1 (L2)
                              const float* __restrict__ scal,
                              const float* __restrict__ xl, const float* __restrict__ xr,
                              const float* __restrict__ We, const float* __restrict__ att,
                              float* __restrict__ logits, unsigned* __restrict__ maxd) {
  int e = blockIdx.x * blockDim.x + threadIdx.x;
  if (e >= Etot) return;
  int s, d;
  edge_sd(ei, E, N, e, s, d);
  float eav;
  if (LAYER == 1) {
    eav = (e < E) ? eaSrc[e] : scal[0] / (float)E;           // mean edge_attr fill
  } else {
    int E1 = Etot - N;                                        // layer-1 edge count
    eav = (e < E1) ? eaSrc[e] : scal[1] / (float)E1;          // mean alpha1 fill
  }
  const float4* xl4 = reinterpret_cast<const float4*>(xl + (size_t)s * C);
  const float4* xr4 = reinterpret_cast<const float4*>(xr + (size_t)d * C);
  const float4* We4 = reinterpret_cast<const float4*>(We);
  const float4* at4 = reinterpret_cast<const float4*>(att);
  float acc = 0.f;
  #pragma unroll
  for (int q = 0; q < C / 4; ++q) {
    float4 a = xl4[q], b = xr4[q], w = We4[q], t = at4[q];
    float v;
    v = a.x + b.x + eav * w.x; v = v > 0.f ? v : NEG_SLOPE * v; acc += v * t.x;
    v = a.y + b.y + eav * w.y; v = v > 0.f ? v : NEG_SLOPE * v; acc += v * t.y;
    v = a.z + b.z + eav * w.z; v = v > 0.f ? v : NEG_SLOPE * v; acc += v * t.z;
    v = a.w + b.w + eav * w.w; v = v > 0.f ? v : NEG_SLOPE * v; acc += v * t.w;
  }
  logits[e] = acc;
  atomicMax(maxd + d, f2ord(acc));
}

// ---- edge pass 2: exp + per-dst denom ------------------------------------
__global__ void edge_exp_k(const int* __restrict__ ei, int E, int N, int Etot,
                           float* __restrict__ logits, const unsigned* __restrict__ maxd,
                           float* __restrict__ sumd) {
  int e = blockIdx.x * blockDim.x + threadIdx.x;
  if (e >= Etot) return;
  int s, d;
  edge_sd(ei, E, N, e, s, d);
  float ex = __expf(logits[e] - ord2f(maxd[d]));
  logits[e] = ex;
  atomicAdd(sumd + d, ex);
}

// ---- edge pass 3: alpha + scatter-add ------------------------------------
template <int C>
__global__ void edge_scatter_k(const int* __restrict__ ei, int E, int N, int Etot,
                               const float* __restrict__ logits, const float* __restrict__ sumd,
                               const float* __restrict__ xl,
                               float* __restrict__ alpha_out, float* __restrict__ out) {
  int e = blockIdx.x * blockDim.x + threadIdx.x;
  if (e >= Etot) return;
  int s, d;
  edge_sd(ei, E, N, e, s, d);
  float alpha = logits[e] / sumd[d];
  alpha_out[e] = alpha;
  const float4* xl4 = reinterpret_cast<const float4*>(xl + (size_t)s * C);
  float* o = out + (size_t)d * C;
  #pragma unroll
  for (int q = 0; q < C / 4; ++q) {
    float4 a = xl4[q];
    atomicAdd(o + 4 * q + 0, alpha * a.x);
    atomicAdd(o + 4 * q + 1, alpha * a.y);
    atomicAdd(o + 4 * q + 2, alpha * a.z);
    atomicAdd(o + 4 * q + 3, alpha * a.w);
  }
}

// ---- ei2 output (src row then dst row, as floats) ------------------------
__global__ void write_ei2_k(const int* __restrict__ ei, int E, int N, int Etot,
                            float* __restrict__ osrc, float* __restrict__ odst) {
  int e = blockIdx.x * blockDim.x + threadIdx.x;
  if (e >= Etot) return;
  int s, d;
  edge_sd(ei, E, N, e, s, d);
  osrc[e] = (float)s;
  odst[e] = (float)d;
}

extern "C" void kernel_launch(void* const* d_in, const int* in_sizes, int n_in,
                              void* d_out, int out_size, void* d_ws, size_t ws_size,
                              hipStream_t stream) {
  (void)n_in; (void)out_size; (void)ws_size;
  const float* x    = (const float*)d_in[0];
  const int*   ei   = (const int*)d_in[1];
  const float* ea   = (const float*)d_in[2];
  const float* W1l  = (const float*)d_in[3];
  const float* W1r  = (const float*)d_in[4];
  const float* W1e  = (const float*)d_in[5];
  const float* att1 = (const float*)d_in[6];
  const float* b1   = (const float*)d_in[7];
  const float* W2l  = (const float*)d_in[8];
  const float* W2r  = (const float*)d_in[9];
  const float* W2e  = (const float*)d_in[10];
  const float* att2 = (const float*)d_in[11];
  const float* b2   = (const float*)d_in[12];

  const int N  = in_sizes[0] / 128;
  const int E  = in_sizes[1] / 2;
  const int E1 = E + N;        // layer-1 edges incl. self loops
  const int E2 = E + 2 * N;    // layer-2 edges (self loops appended again)

  // workspace carve (256B aligned regions)
  uintptr_t p = (uintptr_t)d_ws;
  auto carve = [&](size_t bytes) -> void* {
    p = (p + 255) & ~(uintptr_t)255;
    void* r = (void*)p;
    p += bytes;
    return r;
  };
  float*    scal   = (float*)carve(2 * sizeof(float));   // [0]=sum(ea), [1]=sum(alpha1)
  float*    xl1    = (float*)carve((size_t)N * 16 * 4);
  float*    xr1    = (float*)carve((size_t)N * 16 * 4);
  unsigned* maxd   = (unsigned*)carve((size_t)N * 4);
  float*    sumd   = (float*)carve((size_t)N * 4);
  float*    out1   = (float*)carve((size_t)N * 16 * 4);  // layer-1 out, relu'd in place -> h
  float*    logit1 = (float*)carve((size_t)E1 * 4);
  float*    alpha1 = (float*)carve((size_t)E1 * 4);
  float*    xl2    = (float*)carve((size_t)N * 32 * 4);
  float*    xr2    = (float*)carve((size_t)N * 32 * 4);
  float*    logit2 = (float*)carve((size_t)E2 * 4);

  // output layout: out [N,32] | ei2 src [E2] | ei2 dst [E2] | a2 [E2]
  float* out2  = (float*)d_out;
  float* eisrc = out2 + (size_t)N * 32;
  float* eidst = eisrc + E2;
  float* a2    = eidst + E2;

  const int Ntiles = (N + 15) / 16;
  dim3 blkN((unsigned)((N + 255) / 256));
  dim3 blkE1((unsigned)((E1 + 255) / 256));
  dim3 blkE2((unsigned)((E2 + 255) / 256));
  dim3 blkG((unsigned)((Ntiles + 3) / 4));   // 4 waves / 128-thread block

  // -------- layer 1 --------
  init1_k<<<blkN, 256, 0, stream>>>(maxd, sumd, out1, b1, scal, N);
  reduce_sum_k<<<1024, 256, 0, stream>>>(ea, (long long)E, &scal[0]);
  gemm1_k<<<blkG, 128, 0, stream>>>(x, W1l, W1r, xl1, xr1, Ntiles, N);
  edge_logits_k<16, 1><<<blkE1, 256, 0, stream>>>(ei, E, N, E1, ea, scal,
                                                  xl1, xr1, W1e, att1, logit1, maxd);
  edge_exp_k<<<blkE1, 256, 0, stream>>>(ei, E, N, E1, logit1, maxd, sumd);
  edge_scatter_k<16><<<blkE1, 256, 0, stream>>>(ei, E, N, E1, logit1, sumd, xl1,
                                                alpha1, out1);

  // -------- between layers: relu, re-init, mean(alpha1) --------
  layer1_finish_k<<<blkN, 256, 0, stream>>>(out1, maxd, sumd, out2, b2, scal, N);
  reduce_sum_k<<<1024, 256, 0, stream>>>(alpha1, (long long)E1, &scal[1]);

  // -------- layer 2 --------
  gemm2_k<<<blkG, 128, 0, stream>>>(out1, W2l, W2r, xl2, xr2, Ntiles, N);
  edge_logits_k<32, 2><<<blkE2, 256, 0, stream>>>(ei, E, N, E2, alpha1, scal,
                                                  xl2, xr2, W2e, att2, logit2, maxd);
  edge_exp_k<<<blkE2, 256, 0, stream>>>(ei, E, N, E2, logit2, maxd, sumd);
  edge_scatter_k<32><<<blkE2, 256, 0, stream>>>(ei, E, N, E2, logit2, sumd, xl2,
                                                a2, out2);
  write_ei2_k<<<blkE2, 256, 0, stream>>>(ei, E, N, E2, eisrc, eidst);
}
